// Attention_Embedding_20744692040370
// MI455X (gfx1250) — matrix operand order repaired and checked by reading, not compiled, not run on hardware
//
#include <hip/hip_runtime.h>

typedef __attribute__((ext_vector_type(16))) _Float16 v16h;
typedef __attribute__((ext_vector_type(8)))  float    v8f;

namespace {
constexpr int kB     = 4;
constexpr int kN     = 8100;   // H*W*D = 9*9*100
constexpr int kC     = 64;
constexpr int kNPad  = 8192;   // multiple of 128 (WG row coverage)
constexpr int kJT    = kNPad / 32;   // 256 key tiles
constexpr int kD     = 100;
constexpr int kDo    = 97;     // valid conv outputs along D
constexpr int kHW    = 81;     // 9*9
}

union U16H { v16h v; uint4 u[2]; _Float16 h[16]; };
union U2H  { uint2 u; unsigned short s[4]; };

__device__ __forceinline__ v8f wmma32f16(v16h a, v16h b, v8f c) {
  // 8 args: (neg_a, A, neg_b, B, c_mod, C, reuse_a, reuse_b)
  return __builtin_amdgcn_wmma_f32_16x16x32_f16(false, a, false, b, (short)0, c,
                                                false, false);
}

// ---------------------------------------------------------------------------
// x (f32, zero-padded rows >= kN) -> f16 copy for WMMA operands
// ---------------------------------------------------------------------------
__global__ __launch_bounds__(256)
void k_cvt(const float* __restrict__ x, _Float16* __restrict__ Xh) {
  const size_t i = (size_t)blockIdx.x * 256 + threadIdx.x;
  const size_t total = (size_t)kB * kNPad * kC;
  if (i >= total) return;
  const int    c  = (int)(i % kC);
  const size_t rc = i / kC;
  const int    r  = (int)(rc % kNPad);
  const int    b  = (int)(rc / kNPad);
  float v = 0.f;
  if (r < kN) v = x[((size_t)b * kN + r) * kC + c];
  Xh[i] = (_Float16)v;
}

// ---------------------------------------------------------------------------
// Channel attention: ch = beta * softmax(X X^T) X + x   (flash, WMMA f16)
// grid (kNPad/128, B), block 256 (8 waves; each wave = one 16-row Q tile).
// QK^T B-operands stream from global (VMEM/WGP$ path); only the transposed
// key tile (for PV) and the per-wave P scratch live in LDS.
// ---------------------------------------------------------------------------
__global__ __launch_bounds__(256)
void k_flash_ch(const _Float16* __restrict__ Xh, const float* __restrict__ x,
                const float* __restrict__ betap, float* __restrict__ ch)
{
  __shared__ __align__(16) _Float16 Xt[kC * 32];     // key tile, transposed
  __shared__ __align__(16) _Float16 Pb[8][16 * 32];  // per-wave P scratch

  const int b    = blockIdx.y;
  const int tid  = threadIdx.x;
  const int wave = tid >> 5;
  const int lane = tid & 31;
  const int lh   = lane & 15;   // 0..15
  const int hi   = lane >> 4;   // 0/1 half-group

  const _Float16* Xb = Xh + (size_t)b * kNPad * kC;

  // ---- Q tile A-operands (rows rbase..rbase+15, c chunks [0,32) and [32,64))
  const int rbase = blockIdx.x * 128 + wave * 16;
  U16H a0, a1;
  {
    const _Float16* qp = Xb + (size_t)(rbase + lh) * kC + hi * 16;
    a0.u[0] = *(const uint4*)(qp);        // K 0..7   (hi=0) / 16..23 (hi=1)
    a0.u[1] = *(const uint4*)(qp + 8);    // K 8..15  (hi=0) / 24..31 (hi=1)
    a1.u[0] = *(const uint4*)(qp + 32);
    a1.u[1] = *(const uint4*)(qp + 40);
  }

  U16H onesv;
  #pragma unroll
  for (int i = 0; i < 16; ++i) onesv.h[i] = (_Float16)1.0f;

  const v8f vzero = {0.f, 0.f, 0.f, 0.f, 0.f, 0.f, 0.f, 0.f};
  v8f O[4] = {vzero, vzero, vzero, vzero};   // 16x64 output accumulator
  float M[8], L[8];
  #pragma unroll
  for (int i = 0; i < 8; ++i) { M[i] = -3.0e38f; L[i] = 0.f; }

  const int kp = tid & 15;          // key pair for transposed-tile build
  const int c0 = (tid >> 4) * 4;    // 4-channel slab for transposed-tile build

  for (int jt = 0; jt < kJT; ++jt) {
    const int jb = jt * 32;

    // ---- B operands for S = Q K^T straight from global (B lane l<16 holds
    // column n=l, K 0..15 -> 32 contiguous bytes of key row jb+s*16+l).
    U16H b00, b01, b10, b11;
    {
      const _Float16* kr0 = Xb + (size_t)(jb + lh) * kC + hi * 16;
      const _Float16* kr1 = Xb + (size_t)(jb + 16 + lh) * kC + hi * 16;
      b00.u[0] = *(const uint4*)(kr0);      b00.u[1] = *(const uint4*)(kr0 + 8);
      b01.u[0] = *(const uint4*)(kr0 + 32); b01.u[1] = *(const uint4*)(kr0 + 40);
      b10.u[0] = *(const uint4*)(kr1);      b10.u[1] = *(const uint4*)(kr1 + 8);
      b11.u[0] = *(const uint4*)(kr1 + 32); b11.u[1] = *(const uint4*)(kr1 + 40);
    }

    __syncthreads();   // previous iteration's PV reads of Xt complete
    { // cooperative build of transposed key tile (64 c x 32 keys)
      const _Float16* rp = Xb + (size_t)(jb + 2 * kp) * kC + c0;
      U2H r0, r1;
      r0.u = *(const uint2*)(rp);
      r1.u = *(const uint2*)(rp + kC);
      #pragma unroll
      for (int i = 0; i < 4; ++i) {
        unsigned int pk = (unsigned int)r0.s[i] | ((unsigned int)r1.s[i] << 16);
        *(unsigned int*)&Xt[(c0 + i) * 32 + 2 * kp] = pk;
      }
      if (jt + 1 < kJT) __builtin_prefetch(rp + 32 * kC, 0, 3);
    }
    __syncthreads();

    // ---- S = Q K^T : two 16x16 subtiles over 32 keys, K-dim = C = 64
    v8f S0 = vzero, S1 = vzero;
    S0 = wmma32f16(a0.v, b00.v, S0);
    S0 = wmma32f16(a1.v, b01.v, S0);
    S1 = wmma32f16(a0.v, b10.v, S1);
    S1 = wmma32f16(a1.v, b11.v, S1);

    // mask padded keys (column index = lane's lh in D layout)
    if (jb + lh >= kN) {
      #pragma unroll
      for (int i = 0; i < 8; ++i) S0[i] = -3.0e38f;
    }
    if (jb + 16 + lh >= kN) {
      #pragma unroll
      for (int i = 0; i < 8; ++i) S1[i] = -3.0e38f;
    }

    // ---- online softmax: row max via 16-lane xor reduction
    float rmax[8];
    #pragma unroll
    for (int i = 0; i < 8; ++i) rmax[i] = fmaxf(S0[i], S1[i]);
    #pragma unroll
    for (int off = 8; off > 0; off >>= 1) {
      #pragma unroll
      for (int i = 0; i < 8; ++i)
        rmax[i] = fmaxf(rmax[i], __shfl_xor(rmax[i], off, 32));
    }
    float alpha[8];
    #pragma unroll
    for (int i = 0; i < 8; ++i) {
      const float mn = fmaxf(M[i], rmax[i]);
      alpha[i] = __expf(M[i] - mn);
      M[i] = mn;
    }
    float P0[8], P1[8];
    #pragma unroll
    for (int i = 0; i < 8; ++i) {
      P0[i] = __expf(S0[i] - M[i]);
      P1[i] = __expf(S1[i] - M[i]);
    }

    // ---- P: D-layout f32 -> LDS row-major f16 -> A-layout reload
    _Float16* pb = Pb[wave];
    #pragma unroll
    for (int i = 0; i < 8; ++i) {
      const int m = i + 8 * hi;
      pb[m * 32 + lh]      = (_Float16)P0[i];
      pb[m * 32 + 16 + lh] = (_Float16)P1[i];
    }
    U16H pA;
    {
      const _Float16* pr = pb + lh * 32 + hi * 16;
      pA.u[0] = *(const uint4*)(pr);        // K {0..7} / {16..23}
      pA.u[1] = *(const uint4*)(pr + 8);    // K {8..15} / {24..31}
    }

    // row sums for free via WMMA against all-ones B
    const v8f rs = wmma32f16(pA.v, onesv.v, vzero);
    #pragma unroll
    for (int i = 0; i < 8; ++i) L[i] = L[i] * alpha[i] + rs[i];

    // ---- O = O*alpha + P V   (V columns from transposed LDS tile)
    #pragma unroll
    for (int q = 0; q < 4; ++q) {
      #pragma unroll
      for (int i = 0; i < 8; ++i) O[q][i] *= alpha[i];
      U16H bv;
      const _Float16* vp = &Xt[(q * 16 + lh) * 32 + hi * 16];
      bv.u[0] = *(const uint4*)(vp);
      bv.u[1] = *(const uint4*)(vp + 8);
      O[q] = wmma32f16(pA.v, bv.v, O[q]);
    }
  }

  // ---- epilogue: ch = beta * O/L + x
  const float beta = betap[0];
  float inv[8];
  #pragma unroll
  for (int i = 0; i < 8; ++i) inv[i] = 1.0f / L[i];
  #pragma unroll
  for (int q = 0; q < 4; ++q) {
    #pragma unroll
    for (int i = 0; i < 8; ++i) {
      const int r = rbase + i + 8 * hi;
      if (r < kN) {
        const size_t idx = ((size_t)b * kN + r) * kC + q * 16 + lh;
        ch[idx] = beta * (O[q][i] * inv[i]) + x[idx];
      }
    }
  }
}

// ---------------------------------------------------------------------------
// q/k/v = x @ w + b   grid (127, B), block 256, LDS-tiled
// ---------------------------------------------------------------------------
__global__ __launch_bounds__(256)
void k_qkv(const float* __restrict__ x,
           const float* __restrict__ wq, const float* __restrict__ bq,
           const float* __restrict__ wk, const float* __restrict__ bk,
           const float* __restrict__ wv, const float* __restrict__ bv,
           float* __restrict__ qo, float* __restrict__ ko, float* __restrict__ vo)
{
  __shared__ float xsT[64 * 64];       // [ci][row]
  __shared__ float ww[3][64 * 64];     // [ci][co]
  const int b = blockIdx.y, r0 = blockIdx.x * 64, t = threadIdx.x;
  for (int i = t; i < 4096; i += 256) {
    ww[0][i] = wq[i]; ww[1][i] = wk[i]; ww[2][i] = wv[i];
    const int r = i >> 6, c = i & 63, gr = r0 + r;
    xsT[c * 64 + r] = (gr < kN) ? x[((size_t)b * kN + gr) * kC + c] : 0.f;
  }
  __syncthreads();
  const int co = t & 63, rg = t >> 6;
  const float aq0 = bq[co], ak0 = bk[co], av0 = bv[co];
  for (int ii = 0; ii < 16; ++ii) {
    const int r = rg + 4 * ii;
    float aq = aq0, ak = ak0, av = av0;
    for (int ci = 0; ci < 64; ++ci) {
      const float xv = xsT[ci * 64 + r];
      aq += xv * ww[0][ci * 64 + co];
      ak += xv * ww[1][ci * 64 + co];
      av += xv * ww[2][ci * 64 + co];
    }
    const int gr = r0 + r;
    if (gr < kN) {
      const size_t idx = ((size_t)b * kN + gr) * kC + co;
      qo[idx] = aq; ko[idx] = ak; vo[idx] = av;
    }
  }
}

// ---------------------------------------------------------------------------
// energy_c[b,c,e] += sum_n q[n,c]*k[n,e]   grid (32 slices, B), block 256
// ---------------------------------------------------------------------------
__global__ __launch_bounds__(256)
void k_energy(const float* __restrict__ q, const float* __restrict__ kk,
              float* __restrict__ energy)
{
  __shared__ float qs[64 * 64];
  __shared__ float ks[64 * 64];
  const int b = blockIdx.y, slice = blockIdx.x, t = threadIdx.x;
  const int r0 = slice * 254;
  float acc[16];
  #pragma unroll
  for (int ii = 0; ii < 16; ++ii) acc[ii] = 0.f;
  for (int c0 = 0; c0 < 254; c0 += 64) {
    __syncthreads();
    for (int i = t; i < 4096; i += 256) {
      const int rr = i >> 6, c = i & 63;
      const int gr = r0 + c0 + rr;
      const bool ok = (c0 + rr < 254) && (gr < kN);
      qs[i] = ok ? q[((size_t)b * kN + gr) * kC + c] : 0.f;
      ks[i] = ok ? kk[((size_t)b * kN + gr) * kC + c] : 0.f;
    }
    __syncthreads();
    #pragma unroll
    for (int ii = 0; ii < 16; ++ii) {
      const int p = ii * 256 + t;
      const int c = p >> 6, e = p & 63;
      float a = acc[ii];
      for (int rr = 0; rr < 64; ++rr) a += qs[rr * 64 + c] * ks[rr * 64 + e];
      acc[ii] = a;
    }
  }
  #pragma unroll
  for (int ii = 0; ii < 16; ++ii) {
    const int p = ii * 256 + t;
    const int c = p >> 6, e = p & 63;
    atomicAdd(&energy[((size_t)b * 64 + c) * 64 + e], acc[ii]);
  }
}

// ---------------------------------------------------------------------------
// softmax rows of energy_c   grid (64, B), block 32 (one wave per row)
// ---------------------------------------------------------------------------
__global__ __launch_bounds__(32)
void k_softmax(const float* __restrict__ energy, float* __restrict__ attn)
{
  const int b = blockIdx.y, c = blockIdx.x, l = threadIdx.x;
  const float* E = energy + ((size_t)b * 64 + c) * 64;
  const float e0 = E[l], e1 = E[l + 32];
  float m = fmaxf(e0, e1);
  for (int o = 16; o > 0; o >>= 1) m = fmaxf(m, __shfl_xor(m, o, 32));
  const float p0 = __expf(e0 - m), p1 = __expf(e1 - m);
  float s = p0 + p1;
  for (int o = 16; o > 0; o >>= 1) s += __shfl_xor(s, o, 32);
  float* A = attn + ((size_t)b * 64 + c) * 64;
  A[l] = p0 / s;
  A[l + 32] = p1 / s;
}

// ---------------------------------------------------------------------------
// pos = gamma * (v @ attn^T) + x   grid (507, B), block 256
// ---------------------------------------------------------------------------
__global__ __launch_bounds__(256)
void k_pos(const float* __restrict__ v, const float* __restrict__ attn,
           const float* __restrict__ x, const float* __restrict__ gammap,
           float* __restrict__ pos)
{
  __shared__ float As[64 * 65];
  __shared__ float vs[16 * 65];
  const int b = blockIdx.y, t = threadIdx.x;
  const int r0 = blockIdx.x * 16;
  for (int i = t; i < 4096; i += 256)
    As[(i >> 6) * 65 + (i & 63)] = attn[(size_t)b * 4096 + i];
  for (int i = t; i < 1024; i += 256) {
    const int rr = i >> 6, c = i & 63, gr = r0 + rr;
    vs[rr * 65 + c] = (gr < kN) ? v[((size_t)b * kN + gr) * kC + c] : 0.f;
  }
  __syncthreads();
  const float g = gammap[0];
  const int c = t & 63, rg = t >> 6;
  #pragma unroll
  for (int ii = 0; ii < 4; ++ii) {
    const int rr = rg + 4 * ii;
    float a = 0.f;
    for (int e = 0; e < 64; ++e) a += vs[rr * 65 + e] * As[c * 65 + e];
    const int gr = r0 + rr;
    if (gr < kN) {
      const size_t idx = ((size_t)b * kN + gr) * kC + c;
      pos[idx] = g * a + x[idx];
    }
  }
}

// ---------------------------------------------------------------------------
// out = relu(conv114(ch)+b_ch) + relu(conv114(pos)+b_pos)
// grid (81, B), block 256 — one (h,w) column per block, LDS input slabs
// ---------------------------------------------------------------------------
__global__ __launch_bounds__(256)
void k_conv(const float* __restrict__ chb, const float* __restrict__ posb,
            const float* __restrict__ wch, const float* __restrict__ bch,
            const float* __restrict__ wpos, const float* __restrict__ bpos,
            float* __restrict__ out)
{
  __shared__ float sc[(kD + 3) * 65];
  __shared__ float sp[(kD + 3) * 65];
  const int b = blockIdx.y, hw = blockIdx.x, t = threadIdx.x;
  const size_t base = ((size_t)b * kHW + hw) * kD * kC;
  for (int i = t; i < (kD + 3) * kC; i += 256) {
    const int d = i >> 6, c = i & 63;
    float cv = 0.f, pv = 0.f;
    if (d < kD) { cv = chb[base + i]; pv = posb[base + i]; }
    sc[d * 65 + c] = cv;
    sp[d * 65 + c] = pv;
  }
  __syncthreads();
  const int c = t & 63, dg = t >> 6;
  float a1[25], a2[25];
  const float bc1 = bch[c], bc2 = bpos[c];
  #pragma unroll
  for (int ii = 0; ii < 25; ++ii) { a1[ii] = bc1; a2[ii] = bc2; }
  for (int t4 = 0; t4 < 4; ++t4) {
    for (int ci = 0; ci < 64; ++ci) {
      const float w1 = wch[(size_t)(t4 * 64 + ci) * 64 + c];
      const float w2 = wpos[(size_t)(t4 * 64 + ci) * 64 + c];
      const float* r1 = &sc[t4 * 65 + ci];
      const float* r2 = &sp[t4 * 65 + ci];
      #pragma unroll
      for (int ii = 0; ii < 25; ++ii) {
        const int dd = dg + 4 * ii;
        a1[ii] += r1[dd * 65] * w1;
        a2[ii] += r2[dd * 65] * w2;
      }
    }
  }
  #pragma unroll
  for (int ii = 0; ii < 25; ++ii) {
    const int dd = dg + 4 * ii;
    if (dd < kDo)
      out[(((size_t)b * kHW + hw) * kDo + dd) * kC + c] =
          fmaxf(a1[ii], 0.f) + fmaxf(a2[ii], 0.f);
  }
}

// ---------------------------------------------------------------------------
extern "C" void kernel_launch(void* const* d_in, const int* in_sizes, int n_in,
                              void* d_out, int out_size, void* d_ws, size_t ws_size,
                              hipStream_t stream)
{
  (void)in_sizes; (void)n_in; (void)out_size; (void)ws_size;
  const float* x     = (const float*)d_in[0];
  const float* beta  = (const float*)d_in[1];
  const float* gamma = (const float*)d_in[2];
  const float* wq    = (const float*)d_in[3];
  const float* bq    = (const float*)d_in[4];
  const float* wk    = (const float*)d_in[5];
  const float* bk    = (const float*)d_in[6];
  const float* wv    = (const float*)d_in[7];
  const float* bv    = (const float*)d_in[8];
  const float* wch   = (const float*)d_in[9];
  const float* bch   = (const float*)d_in[10];
  const float* wpos  = (const float*)d_in[11];
  const float* bpos  = (const float*)d_in[12];
  float* out = (float*)d_out;

  // workspace layout (all chunks multiple of 256 bytes); ~44 MB total
  char* ws = (char*)d_ws;
  size_t off = 0;
  _Float16* Xh   = (_Float16*)(ws + off); off += (size_t)kB * kNPad * kC * 2;
  float*    chb  = (float*)(ws + off);    off += (size_t)kB * kN * kC * 4;
  float*    posb = (float*)(ws + off);    off += (size_t)kB * kN * kC * 4;
  float*    qb   = (float*)(ws + off);    off += (size_t)kB * kN * kC * 4;
  float*    kb   = (float*)(ws + off);    off += (size_t)kB * kN * kC * 4;
  float*    vb   = (float*)(ws + off);    off += (size_t)kB * kN * kC * 4;
  float*    enb  = (float*)(ws + off);    off += (size_t)kB * 64 * 64 * 4;
  float*    atb  = (float*)(ws + off);    off += (size_t)kB * 64 * 64 * 4;

  const int cvtTotal = kB * kNPad * kC;
  k_cvt<<<(cvtTotal + 255) / 256, 256, 0, stream>>>(x, Xh);

  k_flash_ch<<<dim3(kNPad / 128, kB), 256, 0, stream>>>(Xh, x, beta, chb);

  k_qkv<<<dim3(127, kB), 256, 0, stream>>>(x, wq, bq, wk, bk, wv, bv, qb, kb, vb);
  hipMemsetAsync(enb, 0, (size_t)kB * 64 * 64 * 4, stream);
  k_energy<<<dim3(32, kB), 256, 0, stream>>>(qb, kb, enb);
  k_softmax<<<dim3(64, kB), 32, 0, stream>>>(enb, atb);
  k_pos<<<dim3(507, kB), 256, 0, stream>>>(vb, atb, x, gamma, posb);

  k_conv<<<dim3(kHW, kB), 256, 0, stream>>>(chb, posb, wch, bch, wpos, bpos, out);
}
